// LocalAttention_12670153523517
// MI455X (gfx1250) — matrix operand order repaired and checked by reading, not compiled, not run on hardware
//
#include <hip/hip_runtime.h>

#define D_MODEL 1536
#define SEQ     2048
#define BATCH   2
#define TOK     (BATCH*SEQ)
#define HEADS   12
#define HD      128
#define NEGINF  (-__builtin_inff())

typedef __attribute__((ext_vector_type(16))) __bf16 v16bf;
typedef __attribute__((ext_vector_type(8)))  float  v8f;
typedef __attribute__((ext_vector_type(4)))  unsigned v4u;
typedef __attribute__((ext_vector_type(8)))  unsigned v8u;

#define ASM_WAIT_ASYNCCNT(n) asm volatile("s_wait_asynccnt " #n ::: "memory")

__device__ __forceinline__ v8f wmma_bf16(v16bf a, v16bf b, v8f c) {
  return __builtin_amdgcn_wmma_f32_16x16x32_bf16(false, a, false, b, (short)0, c, false, false);
}

__device__ __forceinline__ v8f v8f_zero() {
  v8f z;
#pragma unroll
  for (int i = 0; i < 8; ++i) z[i] = 0.0f;
  return z;
}

// Per-lane async DMA: 16 bytes global -> LDS, tracked by ASYNCcnt (no VGPR data).
__device__ __forceinline__ void async_ld16(unsigned lds_byte, const __bf16* g) {
  asm volatile("global_load_async_to_lds_b128 %0, %1, off"
               :: "v"(lds_byte), "v"((unsigned long long)(uintptr_t)g)
               : "memory");
}

// Tensor Data Mover: 2D tile (tile_d0 x tile_d1 elements, 2-byte) global -> LDS.
// D# group0: count=1 | lds_addr | global_addr(57b) | type=2.  Group1 per ISA 8.4.
__device__ __forceinline__ void tdm_load_2d_bf16(unsigned lds_off, const void* gptr,
                                                 unsigned tile_d0, unsigned tile_d1,
                                                 unsigned tensor_d0, unsigned tensor_d1,
                                                 unsigned long long stride0) {
  const unsigned long long ga = (unsigned long long)(uintptr_t)gptr;
  v4u g0; v8u g1;
  g0[0] = 1u;                                               // count=1, user descriptor
  g0[1] = lds_off;                                          // lds_addr [63:32]
  g0[2] = (unsigned)ga;                                     // global_addr lo
  g0[3] = (unsigned)((ga >> 32) & 0x1FFFFFFu) | (2u << 30); // global_addr hi | type=2
  g1[0] = (1u << 16);                                       // data_size=1 (2 bytes)
  g1[1] = (tensor_d0 & 0xFFFFu) << 16;                      // tensor_dim0[15:0] @ bit48
  g1[2] = (tensor_d0 >> 16) | ((tensor_d1 & 0xFFFFu) << 16);
  g1[3] = (tensor_d1 >> 16) | ((tile_d0 & 0xFFFFu) << 16);  // tile_dim0 @ [127:112]
  g1[4] = tile_d1 & 0xFFFFu;                                // tile_dim1; tile_dim2=0
  g1[5] = (unsigned)stride0;                                // tensor_dim0_stride lo32
  g1[6] = (unsigned)(stride0 >> 32) & 0xFFFFu;              // stride hi16; dim1_stride=0
  g1[7] = 0u;
  asm volatile("tensor_load_to_lds %0, %1" :: "s"(g0), "s"(g1) : "memory");
}

// A-fragment: 16x32 bf16, rows M in lanes, K striped per ISA table.
__device__ __forceinline__ v16bf frag_a(const __bf16* base, int pitch, int lane) {
  const int r = lane & 15, h = lane >> 4;
  v16bf a;
#pragma unroll
  for (int p = 0; p < 8; ++p) {
    const int kk = ((p >> 2) << 4) + (h << 3) + ((p & 3) << 1);
    a[2*p]   = base[r * pitch + kk];
    a[2*p+1] = base[r * pitch + kk + 1];
  }
  return a;
}

// B-fragment: 32x16 bf16 (KxN), read from an [N][K]-layout buffer.
__device__ __forceinline__ v16bf frag_b(const __bf16* base, int pitch, int lane) {
  const int c = lane & 15, h = lane >> 4;
  v16bf b;
#pragma unroll
  for (int p = 0; p < 8; ++p) {
    const int kk = ((p >> 2) << 4) + (h << 3) + ((p & 3) << 1);
    b[2*p]   = base[c * pitch + kk];
    b[2*p+1] = base[c * pitch + kk + 1];
  }
  return b;
}

__device__ __forceinline__ float rowred_max(float x) {
#pragma unroll
  for (int off = 1; off < 16; off <<= 1) x = fmaxf(x, __shfl_xor(x, off, 32));
  return x;
}
__device__ __forceinline__ float rowred_sum(float x) {
#pragma unroll
  for (int off = 1; off < 16; off <<= 1) x += __shfl_xor(x, off, 32);
  return x;
}

// ---------------- conversion kernels ----------------
__global__ __launch_bounds__(256) void cvt_x_kernel(const float* __restrict__ x,
                                                    __bf16* __restrict__ xb) {
  const size_t i = (size_t)blockIdx.x * 256 + threadIdx.x;
  xb[i] = (__bf16)x[i];
}

__global__ __launch_bounds__(256) void cvt_w_kernel(
    const float* __restrict__ Wq, const float* __restrict__ Wk,
    const float* __restrict__ Wv, const float* __restrict__ Wo,
    __bf16* __restrict__ wqkvT, __bf16* __restrict__ woT) {
  const size_t i = (size_t)blockIdx.x * 256 + threadIdx.x;
  const size_t DD = (size_t)D_MODEL * D_MODEL;
  const int m = (int)(i / DD);
  const size_t r = i % DD;
  const int n = (int)(r / D_MODEL);
  const int kk = (int)(r % D_MODEL);
  const float* W = (m == 0) ? Wq : ((m == 1) ? Wk : ((m == 2) ? Wv : Wo));
  const __bf16 val = (__bf16)W[(size_t)kk * D_MODEL + n];
  if (m < 3) wqkvT[(size_t)m * DD + (size_t)n * D_MODEL + kk] = val;
  else       woT[(size_t)n * D_MODEL + kk] = val;
}

// ---------------- QKV GEMM: async-DMA double-buffered staging ----------------
__global__ __launch_bounds__(256) void qkv_gemm_kernel(
    const __bf16* __restrict__ xb, const __bf16* __restrict__ wqkvT,
    const float* __restrict__ bq, const float* __restrict__ bk,
    const float* __restrict__ bv,
    __bf16* __restrict__ qout, __bf16* __restrict__ kout, __bf16* __restrict__ vout) {
  __shared__ __bf16 As[2][128 * 32];
  __shared__ __bf16 Bs[2][128 * 32];
  const int tid  = threadIdx.x;
  const int lane = tid & 31, wid = tid >> 5;
  const int wm = wid >> 1, wn = wid & 1;   // 4x2 waves, each 32(M) x 64(N)
  const int t0 = blockIdx.x * 128;
  const int n0 = blockIdx.y * 128;
  const int p  = blockIdx.z;
  const __bf16* wT = wqkvT + (size_t)p * D_MODEL * D_MODEL;

  v8f acc[2][4];
#pragma unroll
  for (int mi = 0; mi < 2; ++mi)
#pragma unroll
    for (int ni = 0; ni < 4; ++ni) acc[mi][ni] = v8f_zero();

  // issue one 128x32 bf16 tile pair into buffer `buf` (4 async ops per wave)
  auto issue_tile = [&](int k0, int buf) {
    const unsigned aB = (unsigned)(uintptr_t)(&As[buf][0]);
    const unsigned bB = (unsigned)(uintptr_t)(&Bs[buf][0]);
#pragma unroll
    for (int inst = 0; inst < 2; ++inst) {
      const int c = tid + inst * 256;          // 512 x 16B chunks per tile
      const int row = c >> 2, o = c & 3;
      async_ld16(aB + row * 64 + o * 16, xb + (size_t)(t0 + row) * D_MODEL + k0 + o * 8);
      async_ld16(bB + row * 64 + o * 16, wT + (size_t)(n0 + row) * D_MODEL + k0 + o * 8);
    }
  };

  issue_tile(0, 0);
  for (int s = 0; s < D_MODEL / 32; ++s) {
    const int k0 = s * 32;
    if (s + 1 < D_MODEL / 32) {
      issue_tile(k0 + 32, (s + 1) & 1);   // prefetch next tile
      ASM_WAIT_ASYNCCNT(4);               // current tile done, next still in flight
    } else {
      ASM_WAIT_ASYNCCNT(0);
    }
    __syncthreads();
    const __bf16* a_t = &As[s & 1][0];
    const __bf16* b_t = &Bs[s & 1][0];
    v16bf af[2], bfr[4];
#pragma unroll
    for (int mi = 0; mi < 2; ++mi) af[mi] = frag_a(a_t + (wm * 32 + mi * 16) * 32, 32, lane);
#pragma unroll
    for (int ni = 0; ni < 4; ++ni) bfr[ni] = frag_b(b_t + (wn * 64 + ni * 16) * 32, 32, lane);
#pragma unroll
    for (int mi = 0; mi < 2; ++mi)
#pragma unroll
      for (int ni = 0; ni < 4; ++ni)
        acc[mi][ni] = wmma_bf16(af[mi], bfr[ni], acc[mi][ni]);
    __syncthreads();   // all waves done reading buf before it is re-filled
  }

  const float scale = (p == 0) ? 0.08838834764831845f : 1.0f;  // 1/sqrt(128) into q
  const float* bias = (p == 0) ? bq : ((p == 1) ? bk : bv);
  __bf16* dst = (p == 0) ? qout : ((p == 1) ? kout : vout);
  const int hh = lane >> 4, cl = lane & 15;
#pragma unroll
  for (int mi = 0; mi < 2; ++mi)
#pragma unroll
    for (int ni = 0; ni < 4; ++ni)
#pragma unroll
      for (int vv = 0; vv < 8; ++vv) {
        const int t = t0 + wm * 32 + mi * 16 + vv + 8 * hh;
        const int n = n0 + wn * 64 + ni * 16 + cl;
        const float val = (acc[mi][ni][vv] + bias[n]) * scale;
        const int head = n >> 7, hd = n & 127;
        const int bb = t >> 11, ss = t & (SEQ - 1);
        dst[(((size_t)bb * HEADS + head) * SEQ + ss) * HD + hd] = (__bf16)val;
      }
}

// ---------------- local-window attention (flash-style online softmax) ----------------
__global__ __launch_bounds__(256) void attn_kernel(
    const __bf16* __restrict__ q, const __bf16* __restrict__ k,
    const __bf16* __restrict__ v, __bf16* __restrict__ attn_out) {
  __shared__ __bf16 Ks[32 * 128];     // [key][hd]
  __shared__ __bf16 Vt[128 * 32];     // [hd][key] (transposed for PV B-fragments)
  __shared__ __bf16 Ps[8][16 * 32];   // per-wave P tile, [row][key32]
  const int tid = threadIdx.x, lane = tid & 31, wid = tid >> 5;
  const int hh = lane >> 4, cl = lane & 15;
  const int q0 = blockIdx.x * 128;
  const int head = blockIdx.y;
  const int b = blockIdx.z;
  const size_t base = (((size_t)b * HEADS + head) * SEQ) * HD;
  const int qrow0 = q0 + wid * 16;

  v16bf aq[4];
  {
    const int r = cl;
#pragma unroll
    for (int ks = 0; ks < 4; ++ks) {
      v16bf a;
#pragma unroll
      for (int p = 0; p < 8; ++p) {
        const int kk = ks * 32 + ((p >> 2) << 4) + (hh << 3) + ((p & 3) << 1);
        a[2*p]   = q[base + (size_t)(qrow0 + r) * HD + kk];
        a[2*p+1] = q[base + (size_t)(qrow0 + r) * HD + kk + 1];
      }
      aq[ks] = a;
    }
  }

  float m[8], l[8];
  v8f o[8];
#pragma unroll
  for (int vv = 0; vv < 8; ++vv) { m[vv] = NEGINF; l[vv] = 0.0f; }
#pragma unroll
  for (int ni = 0; ni < 8; ++ni) o[ni] = v8f_zero();

  const int kstart = q0 - 128;  // window span: 384 keys, 12 tiles of 32
  for (int step = 0; step < 12; ++step) {
    const int kt = kstart + step * 32;
    if (kt + 32 <= 0 || kt >= SEQ) continue;  // uniform across block

    __syncthreads();
#pragma unroll 4
    for (int i = tid; i < 32 * 128; i += 256) {
      const int row = i >> 7, c = i & 127;
      int key = kt + row;
      key = key < 0 ? 0 : (key >= SEQ ? SEQ - 1 : key);
      const size_t ga = base + (size_t)key * HD + c;
      Ks[i] = k[ga];
      Vt[c * 32 + row] = v[ga];
    }
    __syncthreads();

    v8f c0 = v8f_zero(), c1 = v8f_zero();
#pragma unroll
    for (int ks = 0; ks < 4; ++ks)
      c0 = wmma_bf16(aq[ks], frag_b(&Ks[0 * 128 + ks * 32], 128, lane), c0);
#pragma unroll
    for (int ks = 0; ks < 4; ++ks)
      c1 = wmma_bf16(aq[ks], frag_b(&Ks[16 * 128 + ks * 32], 128, lane), c1);

    float pv0[8], pv1[8];
#pragma unroll
    for (int vv = 0; vv < 8; ++vv) {
      const int qi = qrow0 + vv + 8 * hh;
      const int kj0 = kt + cl, kj1 = kt + 16 + cl;
      int d0 = qi - kj0; d0 = d0 < 0 ? -d0 : d0;
      int d1 = qi - kj1; d1 = d1 < 0 ? -d1 : d1;
      pv0[vv] = (kj0 >= 0 && kj0 < SEQ && d0 <= 128) ? c0[vv] : NEGINF;
      pv1[vv] = (kj1 >= 0 && kj1 < SEQ && d1 <= 128) ? c1[vv] : NEGINF;
    }

#pragma unroll
    for (int vv = 0; vv < 8; ++vv) {
      float mx = rowred_max(fmaxf(pv0[vv], pv1[vv]));
      const float mnew = fmaxf(m[vv], mx);
      const float alpha = (m[vv] == mnew) ? 1.0f : __expf(m[vv] - mnew);
      const float p0 = (pv0[vv] == NEGINF) ? 0.0f : __expf(pv0[vv] - mnew);
      const float p1 = (pv1[vv] == NEGINF) ? 0.0f : __expf(pv1[vv] - mnew);
      const float rs = rowred_sum(p0 + p1);
      l[vv] = l[vv] * alpha + rs;
      m[vv] = mnew;
      const int rowi = vv + 8 * hh;
      Ps[wid][rowi * 32 + cl]      = (__bf16)p0;
      Ps[wid][rowi * 32 + 16 + cl] = (__bf16)p1;
#pragma unroll
      for (int ni = 0; ni < 8; ++ni) o[ni][vv] *= alpha;
    }

    const v16bf ap = frag_a(&Ps[wid][0], 32, lane);
#pragma unroll
    for (int ni = 0; ni < 8; ++ni)
      o[ni] = wmma_bf16(ap, frag_b(&Vt[(ni * 16) * 32], 32, lane), o[ni]);
  }

#pragma unroll
  for (int vv = 0; vv < 8; ++vv) {
    const int qi = qrow0 + vv + 8 * hh;
    const float inv = 1.0f / l[vv];
#pragma unroll
    for (int ni = 0; ni < 8; ++ni) {
      const int hd = ni * 16 + cl;
      attn_out[(size_t)(b * SEQ + qi) * D_MODEL + head * HD + hd] =
          (__bf16)(o[ni][vv] * inv);
    }
  }
}

// ---------------- output projection: TDM-staged, double-buffered ----------------
__global__ __launch_bounds__(256) void proj_gemm_kernel(
    const __bf16* __restrict__ attn, const __bf16* __restrict__ woT,
    const float* __restrict__ bo, const float* __restrict__ x,
    float* __restrict__ y) {
  __shared__ __bf16 As[2][128 * 32];
  __shared__ __bf16 Bs[2][128 * 32];
  const int tid = threadIdx.x;
  const int lane = tid & 31, wid = tid >> 5;
  const int wm = wid >> 1, wn = wid & 1;
  const int t0 = blockIdx.x * 128;
  const int n0 = blockIdx.y * 128;

  v8f acc[2][4];
#pragma unroll
  for (int mi = 0; mi < 2; ++mi)
#pragma unroll
    for (int ni = 0; ni < 4; ++ni) acc[mi][ni] = v8f_zero();

  // wave 0 drives the Tensor Data Mover: one D# per 128x32 tile
  auto issue_tdm = [&](int k0, int buf) {
    tdm_load_2d_bf16((unsigned)(uintptr_t)(&As[buf][0]),
                     attn + (size_t)t0 * D_MODEL + k0,
                     32, 128, D_MODEL, 1u << 20, D_MODEL);
    tdm_load_2d_bf16((unsigned)(uintptr_t)(&Bs[buf][0]),
                     woT + (size_t)n0 * D_MODEL + k0,
                     32, 128, D_MODEL, 1u << 20, D_MODEL);
  };

  if (wid == 0) issue_tdm(0, 0);
  for (int s = 0; s < D_MODEL / 32; ++s) {
    const int k0 = s * 32;
    if (wid == 0) {
      if (s + 1 < D_MODEL / 32) {
        issue_tdm(k0 + 32, (s + 1) & 1);                 // prefetch next tile
        __builtin_amdgcn_s_wait_tensorcnt((short)2);     // current tile complete
      } else {
        __builtin_amdgcn_s_wait_tensorcnt((short)0);
      }
    }
    __syncthreads();
    const __bf16* a_t = &As[s & 1][0];
    const __bf16* b_t = &Bs[s & 1][0];
    v16bf af[2], bfr[4];
#pragma unroll
    for (int mi = 0; mi < 2; ++mi) af[mi] = frag_a(a_t + (wm * 32 + mi * 16) * 32, 32, lane);
#pragma unroll
    for (int ni = 0; ni < 4; ++ni) bfr[ni] = frag_b(b_t + (wn * 64 + ni * 16) * 32, 32, lane);
#pragma unroll
    for (int mi = 0; mi < 2; ++mi)
#pragma unroll
      for (int ni = 0; ni < 4; ++ni)
        acc[mi][ni] = wmma_bf16(af[mi], bfr[ni], acc[mi][ni]);
    __syncthreads();
  }

  const int hh = lane >> 4, cl = lane & 15;
#pragma unroll
  for (int mi = 0; mi < 2; ++mi)
#pragma unroll
    for (int ni = 0; ni < 4; ++ni)
#pragma unroll
      for (int vv = 0; vv < 8; ++vv) {
        const int t = t0 + wm * 32 + mi * 16 + vv + 8 * hh;
        const int n = n0 + wn * 64 + ni * 16 + cl;
        y[(size_t)t * D_MODEL + n] =
            acc[mi][ni][vv] + bo[n] + x[(size_t)t * D_MODEL + n];
      }
}

// ---------------- LayerNorm ----------------
__global__ __launch_bounds__(256) void ln_kernel(
    const float* __restrict__ y, const float* __restrict__ gamma,
    const float* __restrict__ beta, float* __restrict__ out) {
  __shared__ float red[256];
  const int tid = threadIdx.x;
  const int row = blockIdx.x;
  const float* yr = y + (size_t)row * D_MODEL;

  float s = 0.0f;
  for (int i = tid; i < D_MODEL; i += 256) s += yr[i];
  red[tid] = s; __syncthreads();
  for (int off = 128; off > 0; off >>= 1) {
    if (tid < off) red[tid] += red[tid + off];
    __syncthreads();
  }
  const float mu = red[0] * (1.0f / D_MODEL);
  __syncthreads();

  float ss = 0.0f;
  for (int i = tid; i < D_MODEL; i += 256) {
    const float d = yr[i] - mu;
    ss += d * d;
  }
  red[tid] = ss; __syncthreads();
  for (int off = 128; off > 0; off >>= 1) {
    if (tid < off) red[tid] += red[tid + off];
    __syncthreads();
  }
  const float rstd = rsqrtf(red[0] * (1.0f / D_MODEL) + 1e-5f);

  float* orow = out + (size_t)row * D_MODEL;
  for (int i = tid; i < D_MODEL; i += 256)
    orow[i] = (yr[i] - mu) * rstd * gamma[i] + beta[i];
}

extern "C" void kernel_launch(void* const* d_in, const int* in_sizes, int n_in,
                              void* d_out, int out_size, void* d_ws, size_t ws_size,
                              hipStream_t stream) {
  (void)in_sizes; (void)n_in; (void)out_size; (void)ws_size;
  const float* x     = (const float*)d_in[0];
  const float* Wq    = (const float*)d_in[1];
  const float* bq    = (const float*)d_in[2];
  const float* Wk    = (const float*)d_in[3];
  const float* bk    = (const float*)d_in[4];
  const float* Wv    = (const float*)d_in[5];
  const float* bv    = (const float*)d_in[6];
  const float* Wo    = (const float*)d_in[7];
  const float* bo    = (const float*)d_in[8];
  const float* gamma = (const float*)d_in[9];
  const float* beta  = (const float*)d_in[10];
  float* out = (float*)d_out;

  char* ws = (char*)d_ws;
  const size_t SZ_TD16 = (size_t)TOK * D_MODEL * sizeof(__bf16);
  const size_t SZ_WQKV = (size_t)3 * D_MODEL * D_MODEL * sizeof(__bf16);
  const size_t SZ_WO   = (size_t)D_MODEL * D_MODEL * sizeof(__bf16);

  __bf16* xb    = (__bf16*)(ws);
  __bf16* wqkvT = (__bf16*)(ws + SZ_TD16);
  float*  y     = (float*)(ws);  // overlaps xb+wqkvT (both dead by proj stage)
  __bf16* woT   = (__bf16*)(ws + SZ_TD16 + SZ_WQKV);
  __bf16* qb    = (__bf16*)(ws + SZ_TD16 + SZ_WQKV + SZ_WO);
  __bf16* kbuf  = qb   + (size_t)TOK * D_MODEL;
  __bf16* vbuf  = kbuf + (size_t)TOK * D_MODEL;
  __bf16* attnb = vbuf + (size_t)TOK * D_MODEL;
  // total workspace footprint ~82 MB

  cvt_x_kernel<<<(TOK * D_MODEL) / 256, 256, 0, stream>>>(x, xb);
  cvt_w_kernel<<<(4 * D_MODEL * D_MODEL) / 256, 256, 0, stream>>>(Wq, Wk, Wv, Wo, wqkvT, woT);
  qkv_gemm_kernel<<<dim3(TOK / 128, D_MODEL / 128, 3), 256, 0, stream>>>(
      xb, wqkvT, bq, bk, bv, qb, kbuf, vbuf);
  attn_kernel<<<dim3(SEQ / 128, HEADS, BATCH), 256, 0, stream>>>(qb, kbuf, vbuf, attnb);
  proj_gemm_kernel<<<dim3(TOK / 128, D_MODEL / 128), 256, 0, stream>>>(attnb, woT, bo, x, y);
  ln_kernel<<<TOK, 256, 0, stream>>>(y, gamma, beta, out);
}
